// SingleGPUMoELayer_48627619725366
// MI455X (gfx1250) — matrix operand order repaired and checked by reading, not compiled, not run on hardware
//
#include <hip/hip_runtime.h>
#include <math.h>

// ---------------- problem constants ----------------
#define T_TOKENS 4096
#define HIDDEN   1024
#define INTER    2816
#define NEXP     8
#define TOTAL_ROWS (T_TOKENS * 2)   // top-2: every token contributes 2 rows

// ---------------- tiling ----------------
#define BM 128
#define BN 64
#define BK 32
#define LDT 40   // padded LDS row stride (bf16 units): 80B/row, breaks bank alignment

typedef __attribute__((ext_vector_type(16))) __bf16 v16bf;
typedef __attribute__((ext_vector_type(2)))  __bf16 v2bf;
typedef __attribute__((ext_vector_type(8)))  float  v8f;
typedef int v4i __attribute__((vector_size(16)));   // matches builtin param type

// -------- CDNA5 async global->LDS path (guarded; fallback = sync copy) ------
#if defined(__has_builtin)
#if __has_builtin(__builtin_amdgcn_global_load_async_to_lds_b128) && \
    __has_builtin(__builtin_amdgcn_s_wait_asynccnt)
#define USE_ASYNC_LDS 1
#endif
#endif
#ifndef USE_ASYNC_LDS
#define USE_ASYNC_LDS 0
#endif

#if USE_ASYNC_LDS
__device__ __forceinline__ void async_cp16(const void* gsrc, void* ldst) {
    __builtin_amdgcn_global_load_async_to_lds_b128(
        (__attribute__((address_space(1))) v4i*)gsrc,
        (__attribute__((address_space(3))) v4i*)ldst,
        0, 0);
}
__device__ __forceinline__ void async_wait0() {
    __builtin_amdgcn_s_wait_asynccnt(0);
}
#endif

// ---------------- bf16 helpers (native hardware converts) ----------------
__device__ __forceinline__ unsigned int cvt2_bf16(float lo, float hi) {
    union { v2bf v; unsigned int u; } c;
    c.v.x = (__bf16)lo;
    c.v.y = (__bf16)hi;
    return c.u;
}
__device__ __forceinline__ unsigned short cvt1_bf16(float f) {
    union { __bf16 b; unsigned short u; } c;
    c.b = (__bf16)f;
    return c.u;
}
// convert 8 consecutive f32 -> 8 bf16, one 16B LDS store
__device__ __forceinline__ void store_b8(unsigned short* dst, float4 a, float4 b) {
    uint4 o;
    o.x = cvt2_bf16(a.x, a.y);
    o.y = cvt2_bf16(a.z, a.w);
    o.z = cvt2_bf16(b.x, b.y);
    o.w = cvt2_bf16(b.z, b.w);
    *(uint4*)dst = o;
}

union FragU { v16bf v; uint4 q[2]; };

// A fragment (16x32 bf16): lane l -> row m=(l&15), k-half kh=(l>>4).
// elements 0..7 = A[m][kh*8+0..7], elements 8..15 = A[m][kh*8+16..23]
__device__ __forceinline__ v16bf lds_frag_a(const unsigned short* base, int m, int kh) {
    FragU u;
    u.q[0] = *(const uint4*)(base + m * LDT + kh * 8);
    u.q[1] = *(const uint4*)(base + m * LDT + kh * 8 + 16);
    return u.v;
}
// B fragment (32x16 bf16, KxN): lane l -> col n=(l&15), k-half kh=(l>>4).
// elements 0..7 = B[n][kh*8+0..7], elements 8..15 = B[n][kh*8+16..23] (weights row-major [N,K])
__device__ __forceinline__ v16bf lds_frag_b(const unsigned short* base, int n, int kh) {
    FragU u;
    const uint4* p = (const uint4*)(base + n * LDT + kh * 8);
    u.q[0] = p[0];
    u.q[1] = p[2];
    return u.v;
}

#define WMMA_BF16(A, B, C) \
    __builtin_amdgcn_wmma_f32_16x16x32_bf16(false, (A), false, (B), (short)0, (C), false, false)

// ---------------- 1. routing: gate GEMV + softmax top-2 ----------------
__global__ __launch_bounds__(256) void moe_route(
    const float* __restrict__ x, const float* __restrict__ gw,
    int* __restrict__ topk_e, float* __restrict__ topk_w, int* __restrict__ counts)
{
    int lane = threadIdx.x & 31;
    int wid  = threadIdx.x >> 5;
    int t = blockIdx.x * 8 + wid;           // one wave32 per token
    float acc[NEXP];
#pragma unroll
    for (int e = 0; e < NEXP; e++) acc[e] = 0.f;
    const float* xr = x + (size_t)t * HIDDEN;
    for (int h = lane; h < HIDDEN; h += 32) {
        float xv = xr[h];
#pragma unroll
        for (int e = 0; e < NEXP; e++) acc[e] += xv * gw[e * HIDDEN + h];
    }
#pragma unroll
    for (int off = 16; off > 0; off >>= 1) {
#pragma unroll
        for (int e = 0; e < NEXP; e++) acc[e] += __shfl_xor(acc[e], off, 32);
    }
    if (lane == 0) {
        int i0 = 0; float l0 = acc[0];
#pragma unroll
        for (int e = 1; e < NEXP; e++) if (acc[e] > l0) { l0 = acc[e]; i0 = e; }
        int i1 = (i0 == 0) ? 1 : 0; float l1 = acc[i1];
#pragma unroll
        for (int e = 0; e < NEXP; e++)
            if (e != i0 && e != i1 && acc[e] > l1) { l1 = acc[e]; i1 = e; }
        // renormalized top-2 softmax: full denominator cancels
        float p1  = expf(l1 - l0);
        float inv = 1.f / (1.f + p1);
        topk_e[t * 2 + 0] = i0;  topk_e[t * 2 + 1] = i1;
        topk_w[t * 2 + 0] = inv; topk_w[t * 2 + 1] = p1 * inv;
        atomicAdd(&counts[i0], 1);
        atomicAdd(&counts[i1], 1);
    }
}

// ---------------- 2. tiny exclusive scan over 8 expert counts ----------------
__global__ void moe_scan(const int* __restrict__ counts, int* __restrict__ offsets)
{
    if (threadIdx.x == 0 && blockIdx.x == 0) {
        int s = 0;
        for (int e = 0; e < NEXP; e++) { offsets[e] = s; s += counts[e]; }
    }
}

// ---------------- 3. build per-expert contiguous row lists ----------------
__global__ __launch_bounds__(256) void moe_build(
    const int* __restrict__ topk_e, const float* __restrict__ topk_w,
    const int* __restrict__ offsets, int* __restrict__ fill,
    int* __restrict__ tok_of_row, float* __restrict__ w_of_row)
{
    int i = blockIdx.x * 256 + threadIdx.x;   // (token, k) pair id
    if (i < TOTAL_ROWS) {
        int e = topk_e[i];
        float w = topk_w[i];
        int pos = atomicAdd(&fill[e], 1);
        int row = offsets[e] + pos;
        tok_of_row[row] = i >> 1;
        w_of_row[row]   = w;
    }
}

// ---------------- 4. x: f32 -> bf16 copy ----------------
__global__ __launch_bounds__(256) void cvt_f32_bf16(
    const float* __restrict__ src, unsigned short* __restrict__ dst, int n4)
{
    int i = blockIdx.x * 256 + threadIdx.x;
    if (i < n4) {
        float4 v = ((const float4*)src)[i];
        uint2 o;
        o.x = cvt2_bf16(v.x, v.y);
        o.y = cvt2_bf16(v.z, v.w);
        ((uint2*)dst)[i] = o;
    }
}

// ---------------- 5. grouped GEMM1 + fused SwiGLU ----------------
// hidden[row, n] = silu(Xg @ w1[e]^T) * (Xg @ w3[e]^T), Xg = gathered bf16 x rows
__global__ __launch_bounds__(256) void moe_gemm1(
    const unsigned short* __restrict__ xb,
    const float* __restrict__ w1, const float* __restrict__ w3,
    const int* __restrict__ counts, const int* __restrict__ offsets,
    const int* __restrict__ tok_of_row,
    unsigned short* __restrict__ hiddenb)
{
    int e  = blockIdx.z;
    int Ne = counts[e];
    int m0 = blockIdx.y * BM;
    if (m0 >= Ne) return;
    int base = offsets[e];
    int n0   = blockIdx.x * BN;

    __shared__ __align__(16) unsigned short As [2][BM * LDT];
    __shared__ __align__(16) unsigned short Bs1[2][BN * LDT];
    __shared__ __align__(16) unsigned short Bs3[2][BN * LDT];
    __shared__ int ids[BM];

    int tid = threadIdx.x;
    if (tid < BM)
        ids[tid] = (m0 + tid < Ne) ? tok_of_row[base + m0 + tid] : 0;
    __syncthreads();

    const float* w1e = w1 + (size_t)e * INTER * HIDDEN;
    const float* w3e = w3 + (size_t)e * INTER * HIDDEN;

    int lane = tid & 31, wid = tid >> 5;
    int wm = (wid & 3) * 32;      // wave M offset in block tile
    int wn = (wid >> 2) * 32;     // wave N offset in block tile
    int fm = lane & 15, kh = lane >> 4;

    // loader coords: A = 128 rows x 2 half-rows, B = 64 rows x 4 chunks of 8
    int ar = tid >> 1, ah = tid & 1;
    int bn = tid >> 2, bq = tid & 3;

    const unsigned short* agp = xb + (size_t)ids[ar] * HIDDEN + ah * 16;
    const float* b1p = w1e + (size_t)(n0 + bn) * HIDDEN + bq * 8;
    const float* b3p = w3e + (size_t)(n0 + bn) * HIDDEN + bq * 8;
    unsigned short* aDst = &As[0][0] + ar * LDT + ah * 16;
    unsigned short* b1Dst = &Bs1[0][0] + bn * LDT + bq * 8;
    unsigned short* b3Dst = &Bs3[0][0] + bn * LDT + bq * 8;
    const int bufStrideA = BM * LDT, bufStrideB = BN * LDT;

    v8f acc1[2][2] = {{{}, {}}, {{}, {}}};
    v8f acc3[2][2] = {{{}, {}}, {{}, {}}};

    // ---- prologue: stage tile 0 ----
    float4 f1a, f1b, f3a, f3b;
#if USE_ASYNC_LDS
    async_cp16(agp,     aDst);
    async_cp16(agp + 8, aDst + 8);
#else
    uint4 a0r = *(const uint4*)agp;
    uint4 a1r = *(const uint4*)(agp + 8);
#endif
    f1a = *(const float4*)b1p; f1b = *(const float4*)(b1p + 4);
    f3a = *(const float4*)b3p; f3b = *(const float4*)(b3p + 4);
    store_b8(b1Dst, f1a, f1b);
    store_b8(b3Dst, f3a, f3b);
#if USE_ASYNC_LDS
    async_wait0();
#else
    *(uint4*)aDst = a0r; *(uint4*)(aDst + 8) = a1r;
#endif
    __syncthreads();

    const int NK = HIDDEN / BK;
    for (int t = 0; t < NK; ++t) {
        int cur = t & 1, nxt = cur ^ 1;
        bool more = (t + 1 < NK);
        if (more) {
            int kc = (t + 1) * BK;
#if USE_ASYNC_LDS
            async_cp16(agp + kc,     aDst + nxt * bufStrideA);
            async_cp16(agp + kc + 8, aDst + nxt * bufStrideA + 8);
#else
            a0r = *(const uint4*)(agp + kc);
            a1r = *(const uint4*)(agp + kc + 8);
#endif
            f1a = *(const float4*)(b1p + kc); f1b = *(const float4*)(b1p + kc + 4);
            f3a = *(const float4*)(b3p + kc); f3b = *(const float4*)(b3p + kc + 4);
            __builtin_prefetch(b1p + kc + BK, 0, 1);
            __builtin_prefetch(b3p + kc + BK, 0, 1);
        }

        const unsigned short* Ac  = &As [cur][0];
        const unsigned short* B1c = &Bs1[cur][0];
        const unsigned short* B3c = &Bs3[cur][0];
        v16bf a0  = lds_frag_a(Ac,  wm + fm,      kh);
        v16bf a1  = lds_frag_a(Ac,  wm + 16 + fm, kh);
        v16bf b10 = lds_frag_b(B1c, wn + fm,      kh);
        v16bf b11 = lds_frag_b(B1c, wn + 16 + fm, kh);
        v16bf b30 = lds_frag_b(B3c, wn + fm,      kh);
        v16bf b31 = lds_frag_b(B3c, wn + 16 + fm, kh);

        acc1[0][0] = WMMA_BF16(a0, b10, acc1[0][0]);
        acc1[0][1] = WMMA_BF16(a0, b11, acc1[0][1]);
        acc1[1][0] = WMMA_BF16(a1, b10, acc1[1][0]);
        acc1[1][1] = WMMA_BF16(a1, b11, acc1[1][1]);
        acc3[0][0] = WMMA_BF16(a0, b30, acc3[0][0]);
        acc3[0][1] = WMMA_BF16(a0, b31, acc3[0][1]);
        acc3[1][0] = WMMA_BF16(a1, b30, acc3[1][0]);
        acc3[1][1] = WMMA_BF16(a1, b31, acc3[1][1]);

        if (more) {
            store_b8(b1Dst + nxt * bufStrideB, f1a, f1b);
            store_b8(b3Dst + nxt * bufStrideB, f3a, f3b);
#if USE_ASYNC_LDS
            async_wait0();
#else
            *(uint4*)(aDst + nxt * bufStrideA)     = a0r;
            *(uint4*)(aDst + nxt * bufStrideA + 8) = a1r;
#endif
        }
        __syncthreads();
    }

    // fused SwiGLU epilogue -> bf16 hidden rows (fast silu: exp + rcp)
#pragma unroll
    for (int i = 0; i < 2; i++)
#pragma unroll
        for (int j = 0; j < 2; j++)
#pragma unroll
            for (int r = 0; r < 8; r++) {
                int mm = wm + i * 16 + kh * 8 + r;    // C layout: m=(lane>>4)*8+vgpr
                if (m0 + mm < Ne) {
                    float h1 = acc1[i][j][r];
                    float h3 = acc3[i][j][r];
                    float hv = h1 * __builtin_amdgcn_rcpf(1.f + __expf(-h1)) * h3;
                    int nn = n0 + wn + j * 16 + fm;   // C layout: n=(lane&15)
                    hiddenb[(size_t)(base + m0 + mm) * INTER + nn] = cvt1_bf16(hv);
                }
            }
}

// ---------------- 6. grouped GEMM2 + weighted scatter ----------------
// out[tok] += w_of_row[row] * (hidden[row] @ w2[e]^T)
__global__ __launch_bounds__(256) void moe_gemm2(
    const unsigned short* __restrict__ hiddenb,
    const float* __restrict__ w2,
    const int* __restrict__ counts, const int* __restrict__ offsets,
    const int* __restrict__ tok_of_row, const float* __restrict__ w_of_row,
    float* __restrict__ out)
{
    int e  = blockIdx.z;
    int Ne = counts[e];
    int m0 = blockIdx.y * BM;
    if (m0 >= Ne) return;
    int base = offsets[e];
    int n0   = blockIdx.x * BN;

    __shared__ __align__(16) unsigned short As[2][BM * LDT];
    __shared__ __align__(16) unsigned short Bs[2][BN * LDT];
    __shared__ int   toks[BM];
    __shared__ float wgts[BM];

    int tid = threadIdx.x;
    if (tid < BM) {
        int ok  = (m0 + tid < Ne);
        int row = base + m0 + tid;
        toks[tid] = ok ? tok_of_row[row] : 0;
        wgts[tid] = ok ? w_of_row[row] : 0.f;
    }
    __syncthreads();

    const float* w2e = w2 + (size_t)e * HIDDEN * INTER;

    int lane = tid & 31, wid = tid >> 5;
    int wm = (wid & 3) * 32;
    int wn = (wid >> 2) * 32;
    int fm = lane & 15, kh = lane >> 4;

    int ar = tid >> 1, ah = tid & 1;
    int bn = tid >> 2, bq = tid & 3;

    int arow = base + m0 + ar;
    if (arow >= TOTAL_ROWS) arow = TOTAL_ROWS - 1;   // clamp (results discarded)
    const unsigned short* agp = hiddenb + (size_t)arow * INTER + ah * 16;
    const float* bp = w2e + (size_t)(n0 + bn) * INTER + bq * 8;
    unsigned short* aDst = &As[0][0] + ar * LDT + ah * 16;
    unsigned short* bDst = &Bs[0][0] + bn * LDT + bq * 8;
    const int bufStrideA = BM * LDT, bufStrideB = BN * LDT;

    v8f acc[2][2] = {{{}, {}}, {{}, {}}};

    // ---- prologue: stage tile 0 ----
    float4 fa, fb;
#if USE_ASYNC_LDS
    async_cp16(agp,     aDst);
    async_cp16(agp + 8, aDst + 8);
#else
    uint4 a0r = *(const uint4*)agp;
    uint4 a1r = *(const uint4*)(agp + 8);
#endif
    fa = *(const float4*)bp; fb = *(const float4*)(bp + 4);
    store_b8(bDst, fa, fb);
#if USE_ASYNC_LDS
    async_wait0();
#else
    *(uint4*)aDst = a0r; *(uint4*)(aDst + 8) = a1r;
#endif
    __syncthreads();

    const int NK = INTER / BK;
    for (int t = 0; t < NK; ++t) {
        int cur = t & 1, nxt = cur ^ 1;
        bool more = (t + 1 < NK);
        if (more) {
            int kc = (t + 1) * BK;
#if USE_ASYNC_LDS
            async_cp16(agp + kc,     aDst + nxt * bufStrideA);
            async_cp16(agp + kc + 8, aDst + nxt * bufStrideA + 8);
#else
            a0r = *(const uint4*)(agp + kc);
            a1r = *(const uint4*)(agp + kc + 8);
#endif
            fa = *(const float4*)(bp + kc); fb = *(const float4*)(bp + kc + 4);
            __builtin_prefetch(bp + kc + BK, 0, 1);
        }

        const unsigned short* Ac = &As[cur][0];
        const unsigned short* Bc = &Bs[cur][0];
        v16bf a0 = lds_frag_a(Ac, wm + fm,      kh);
        v16bf a1 = lds_frag_a(Ac, wm + 16 + fm, kh);
        v16bf b0 = lds_frag_b(Bc, wn + fm,      kh);
        v16bf b1 = lds_frag_b(Bc, wn + 16 + fm, kh);

        acc[0][0] = WMMA_BF16(a0, b0, acc[0][0]);
        acc[0][1] = WMMA_BF16(a0, b1, acc[0][1]);
        acc[1][0] = WMMA_BF16(a1, b0, acc[1][0]);
        acc[1][1] = WMMA_BF16(a1, b1, acc[1][1]);

        if (more) {
            store_b8(bDst + nxt * bufStrideB, fa, fb);
#if USE_ASYNC_LDS
            async_wait0();
#else
            *(uint4*)(aDst + nxt * bufStrideA)     = a0r;
            *(uint4*)(aDst + nxt * bufStrideA + 8) = a1r;
#endif
        }
        __syncthreads();
    }

#pragma unroll
    for (int i = 0; i < 2; i++)
#pragma unroll
        for (int j = 0; j < 2; j++)
#pragma unroll
            for (int r = 0; r < 8; r++) {
                int mm = wm + i * 16 + kh * 8 + r;
                if (m0 + mm < Ne) {
                    float v = acc[i][j][r] * wgts[mm];
                    int nn = n0 + wn + j * 16 + fm;
                    // exactly 2 commutative f32 adds per out element -> deterministic
                    atomicAdd(&out[(size_t)toks[mm] * HIDDEN + nn], v);
                }
            }
}

// ---------------- launch ----------------
extern "C" void kernel_launch(void* const* d_in, const int* in_sizes, int n_in,
                              void* d_out, int out_size, void* d_ws, size_t ws_size,
                              hipStream_t stream)
{
    const float* x      = (const float*)d_in[0];
    const float* gate_w = (const float*)d_in[1];
    const float* w1     = (const float*)d_in[2];
    const float* w2     = (const float*)d_in[3];
    const float* w3     = (const float*)d_in[4];
    float* out = (float*)d_out;
    (void)out;

    char* ws = (char*)d_ws;
    size_t off = 0;
    auto walloc = [&](size_t bytes) -> char* {
        char* p = ws + off;
        off += (bytes + 255) & ~(size_t)255;
        return p;
    };
    int*   counts     = (int*)  walloc(NEXP * 4);          // slot 0
    int*   fill       = (int*)  walloc(NEXP * 4);          // slot 1
    int*   offsets    = (int*)  walloc(NEXP * 4);
    int*   topk_e     = (int*)  walloc(TOTAL_ROWS * 4);
    float* topk_w     = (float*)walloc(TOTAL_ROWS * 4);
    int*   tok_of_row = (int*)  walloc(TOTAL_ROWS * 4);
    float* w_of_row   = (float*)walloc(TOTAL_ROWS * 4);
    unsigned short* xb      = (unsigned short*)walloc((size_t)T_TOKENS * HIDDEN * 2);
    unsigned short* hiddenb = (unsigned short*)walloc((size_t)TOTAL_ROWS * INTER * 2);

    // zero counts + fill (first two 256B slots) and the output accumulator
    (void)hipMemsetAsync(counts, 0, 512, stream);
    (void)hipMemsetAsync(d_out, 0, (size_t)T_TOKENS * HIDDEN * sizeof(float), stream);

    moe_route<<<T_TOKENS / 8, 256, 0, stream>>>(x, gate_w, topk_e, topk_w, counts);
    moe_scan <<<1, 32, 0, stream>>>(counts, offsets);
    moe_build<<<TOTAL_ROWS / 256, 256, 0, stream>>>(topk_e, topk_w, offsets, fill,
                                                    tok_of_row, w_of_row);
    cvt_f32_bf16<<<(T_TOKENS * HIDDEN / 4 + 255) / 256, 256, 0, stream>>>(
        x, xb, T_TOKENS * HIDDEN / 4);

    moe_gemm1<<<dim3(INTER / BN, T_TOKENS / BM, NEXP), 256, 0, stream>>>(
        xb, w1, w3, counts, offsets, tok_of_row, hiddenb);
    moe_gemm2<<<dim3(HIDDEN / BN, T_TOKENS / BM, NEXP), 256, 0, stream>>>(
        hiddenb, w2, counts, offsets, tok_of_row, w_of_row, out);
}
